// Encoder_56238301774532
// MI455X (gfx1250) — matrix operand order repaired and checked by reading, not compiled, not run on hardware
//
#include <hip/hip_runtime.h>

typedef unsigned short ushort_t;
typedef __attribute__((ext_vector_type(16))) __bf16 v16bf;
typedef __attribute__((ext_vector_type(8)))  float  v8f;

union BfFrag { v16bf v; ushort_t u[16]; uint4 q[2]; };

__device__ __forceinline__ ushort_t f2bf(float f) {
  unsigned int u = __float_as_uint(f);
  u += 0x7FFFu + ((u >> 16) & 1u);          // round-to-nearest-even
  return (ushort_t)(u >> 16);
}

// ---------------------------------------------------------------------------
// Generic bf16 WMMA GEMM: C[M x Nout] = act(A[M x K] * B[K x Nout] + bias)
// A: row-major bf16 (AF32==0) or f32 (AF32==1, any lda); B given transposed
// (Bt[Nout x K], row = output column). One wave computes a 16x64 strip with
// four v_wmma_f32_16x16x32_bf16 per K-step (A fragment reused 4x; all four
// B fragments loaded before the WMMA burst so loads clause together).
// ACT:  0=none, 1=relu, 2=leaky_relu(0.01)
// MODE: 0=store f32 to outF, 1=store bf16 to outB, 2=atomicAdd f32 (split-K,
//       grid.y = kchunks, bias/act must be disabled by caller)
// ---------------------------------------------------------------------------
template <int AF32, int ACT, int MODE>
__global__ void gemm_bf16_wmma(const ushort_t* __restrict__ Abf,
                               const float*    __restrict__ Af32,
                               long lda,
                               const ushort_t* __restrict__ Bt, long ldb,
                               const float* __restrict__ bias,
                               float* __restrict__ outF,
                               ushort_t* __restrict__ outB,
                               int Mtiles, int Ntiles64, int K, int Nout,
                               int kchunks)
{
  int wid = (int)((blockIdx.x * blockDim.x + threadIdx.x) >> 5);
  if (wid >= Mtiles * Ntiles64) return;        // wave-uniform exit
  int lane = threadIdx.x & 31;
  int tm = wid / Ntiles64, tn = wid % Ntiles64;
  int half = lane >> 4;                        // 0 or 1
  int l15  = lane & 15;
  long aoff  = (long)(tm * 16 + l15) * lda + half * 8;
  long bbase = (long)(tn * 64 + l15) * ldb + half * 8;
  int Kc   = ((K / kchunks) + 31) & ~31;
  int kbeg = blockIdx.y * Kc;
  int kend = min(K, kbeg + Kc);
  v8f acc[4] = {};
  for (int k0 = kbeg; k0 < kend; k0 += 32) {
    BfFrag a, b[4];
    #pragma unroll
    for (int j = 0; j < 4; ++j) {              // clause all B loads first
      const ushort_t* bp = Bt + bbase + (long)j * 16 * ldb + k0;
      b[j].q[0] = *(const uint4*)(bp);         // K = kb .. kb+7
      b[j].q[1] = *(const uint4*)(bp + 16);    // K = kb+16 .. kb+23
      __builtin_prefetch((const void*)(bp + 32), 0, 1);
    }
    if (AF32) {                                // on-the-fly f32 -> bf16
      const float* ap = Af32 + aoff + k0;
      #pragma unroll
      for (int t = 0; t < 8; ++t) {
        a.u[t]     = f2bf(ap[t]);
        a.u[t + 8] = f2bf(ap[t + 16]);
      }
    } else {
      const ushort_t* ap = Abf + aoff + k0;
      a.q[0] = *(const uint4*)(ap);            // K = kb .. kb+7
      a.q[1] = *(const uint4*)(ap + 16);       // K = kb+16 .. kb+23
      __builtin_prefetch((const void*)(ap + 32), 0, 1);
    }
    #pragma unroll
    for (int j = 0; j < 4; ++j)
      acc[j] = __builtin_amdgcn_wmma_f32_16x16x32_bf16(false, a.v, false,
                                                       b[j].v, (short)0,
                                                       acc[j], false, false);
  }
  int rbase = tm * 16 + half * 8;
  #pragma unroll
  for (int j = 0; j < 4; ++j) {
    int bcol = tn * 64 + j * 16 + l15;
    float bval = bias ? bias[bcol] : 0.0f;
    #pragma unroll
    for (int r = 0; r < 8; ++r) {
      float v = acc[j][r] + bval;
      if (ACT == 1)      v = fmaxf(v, 0.0f);
      else if (ACT == 2) v = (v > 0.0f) ? v : 0.01f * v;
      long o = (long)(rbase + r) * Nout + bcol;
      if (MODE == 0)      outF[o] = v;
      else if (MODE == 1) outB[o] = f2bf(v);
      else                atomicAdd(&outF[o], v);
    }
  }
}

// --------------------- f32 -> bf16 convert ---------------------------------
__global__ void k_cvt(const float* __restrict__ in, ushort_t* out, long total)
{
  long idx = (long)blockIdx.x * blockDim.x + threadIdx.x;
  if (idx < total) out[idx] = f2bf(in[idx]);
}

// --------------------- segment max / count over edges ----------------------
__global__ void k_seg(const int* src, const float* ts, unsigned int* tmax_bits,
                      int* cnt, int E)
{
  int e = blockIdx.x * blockDim.x + threadIdx.x;
  if (e >= E) return;
  int s = src[e];
  atomicMax(&tmax_bits[s], __float_as_uint(ts[e]));   // ts >= 0: bit order ok
  atomicAdd(&cnt[s], 1);
}

// --------------------- edge scatter: agg[N,3,257] --------------------------
__global__ void k_edge_scatter(const int* src, const int* tgt, const float* ts,
                               const float* __restrict__ node_features,
                               const float* __restrict__ memory,
                               const unsigned int* tmax_bits,
                               const float* lambs, float* agg, int E)
{
  int wid = (int)((blockIdx.x * blockDim.x + threadIdx.x) >> 5);
  if (wid >= E) return;
  int lane = threadIdx.x & 31;
  int s = src[wid], t = tgt[wid];
  float d = __uint_as_float(tmax_bits[s]) - ts[wid];
  float w0 = expf(-lambs[0] * d);
  float w1 = expf(-lambs[1] * d);
  float w2 = expf(-lambs[2] * d);
  const float* memt = memory + (long)t * 195;     // (3,65)
  const float* nf   = node_features + (long)t * 64;
  float* aggs = agg + (long)s * 771;              // (3,257)
  for (int k = lane; k < 257; k += 32) {
    float val;
    if (k < 192) { int l2 = k >> 6, j = k & 63; val = memt[l2*65 + j] / memt[l2*65 + 64]; }
    else if (k < 256) val = nf[k - 192];
    else val = 1.0f;
    atomicAdd(aggs +       k, w0 * val);
    atomicAdd(aggs + 257 + k, w1 * val);
    atomicAdd(aggs + 514 + k, w2 * val);
  }
}

// ------------- softmax over axis 0 -> At[r, n] bf16 (masked) ---------------
__global__ void k_col_softmax(const float* __restrict__ A_r, const int* cnt,
                              ushort_t* At, int N, int R, int Npad)
{
  __shared__ float red[256];
  int r = blockIdx.x, tid = threadIdx.x;
  float mx = -3.4e38f;
  for (int n = tid; n < N; n += 256) mx = fmaxf(mx, A_r[(long)n * R + r]);
  red[tid] = mx; __syncthreads();
  for (int s = 128; s > 0; s >>= 1) { if (tid < s) red[tid] = fmaxf(red[tid], red[tid+s]); __syncthreads(); }
  float cmax = red[0]; __syncthreads();
  float sum = 0.f;
  for (int n = tid; n < N; n += 256) sum += expf(A_r[(long)n * R + r] - cmax);
  red[tid] = sum; __syncthreads();
  for (int s = 128; s > 0; s >>= 1) { if (tid < s) red[tid] += red[tid+s]; __syncthreads(); }
  float inv = 1.0f / red[0];
  for (int n = tid; n < Npad; n += 256) {
    float v = 0.f;
    if (n < N && cnt[n] > 0) v = expf(A_r[(long)n * R + r] - cmax) * inv;
    At[(long)r * Npad + n] = f2bf(v);
  }
}

// ------------- softmax over axis 1 -> r2n[n, r] bf16 -----------------------
__global__ void k_row_softmax(const float* __restrict__ A_r, ushort_t* r2n,
                              int N, int R, int Rpad)
{
  int wid = (int)((blockIdx.x * blockDim.x + threadIdx.x) >> 5);
  if (wid >= N) return;
  int lane = threadIdx.x & 31;
  const float* row = A_r + (long)wid * R;
  float mx = -3.4e38f;
  for (int k = lane; k < R; k += 32) mx = fmaxf(mx, row[k]);
  for (int o = 16; o > 0; o >>= 1) mx = fmaxf(mx, __shfl_xor(mx, o, 32));
  float sum = 0.f;
  for (int k = lane; k < R; k += 32) sum += expf(row[k] - mx);
  for (int o = 16; o > 0; o >>= 1) sum += __shfl_xor(sum, o, 32);
  float inv = 1.0f / sum;
  for (int k = lane; k < Rpad; k += 32)
    r2n[(long)wid * Rpad + k] = f2bf(k < R ? expf(row[k] - mx) * inv : 0.f);
}

// ------------- memory decay/update -> node_emb into dyn rows [0,N) ---------
__global__ void k_mem_update(const float* __restrict__ memory,
                             const float* __restrict__ agg,
                             const float* __restrict__ msg_feat,
                             const unsigned int* tmax_bits, const int* cnt,
                             const float* last_update, const float* lambs,
                             ushort_t* dyn_bf, int N)
{
  int idx = blockIdx.x * blockDim.x + threadIdx.x;
  if (idx >= N * 3) return;
  int n = idx / 3, l = idx % 3;
  bool hm = cnt[n] > 0;
  float dt = hm ? (__uint_as_float(tmax_bits[n]) - last_update[n]) : 0.f;
  float dec = expf(-lambs[l] * dt);
  const float* mem = memory   + (long)idx * 65;
  const float* ag  = agg      + (long)idx * 257;
  const float* mf  = msg_feat + (long)idx * 64;
  float nd  = hm ? (mem[64] * dec + ag[256]) : mem[64];
  float dinv = 1.0f / fmaxf(nd, 1e-6f);
  ushort_t* d = dyn_bf + (long)n * 192 + l * 64;
  for (int j = 0; j < 64; ++j) {
    float nm = hm ? (mem[j] * dec + mf[j]) : mem[j];
    d[j] = f2bf(nm * dinv);
  }
}

// ------------- B matrix, transposed -> Bt[768 x Npad] bf16 -----------------
__global__ void k_postB(const float* __restrict__ agg, const int* cnt,
                        ushort_t* Bt, int N, int Npad)
{
  long idx = (long)blockIdx.x * blockDim.x + threadIdx.x;
  if (idx >= (long)768 * Npad) return;
  int f = (int)(idx / Npad), n = (int)(idx % Npad);
  float v = 0.f;
  if (n < N && cnt[n] > 0) {
    int l = f >> 8, c = f & 255;
    const float* ag = agg + ((long)n * 3 + l) * 257;
    v = ag[c] / fmaxf(ag[256], 1e-6f);
  }
  Bt[idx] = f2bf(v);
}

// ------------- virtual memory update -> v_embT[192 x Rpad] bf16 ------------
__global__ void k_vupdate(const float* __restrict__ vmem, const float* vlu,
                          const float* now, const float* lambs,
                          const float* __restrict__ vmsg,
                          ushort_t* vembT, int R, int Rpad)
{
  int idx = blockIdx.x * blockDim.x + threadIdx.x;
  if (idx >= 192 * Rpad) return;
  int f = idx / Rpad, r = idx % Rpad;
  float v = 0.f;
  if (r < R) {
    int l = f >> 6, j = f & 63;
    float vdec = expf(-lambs[l] * (now[0] - vlu[r]));
    v = vmem[((long)r * 3 + l) * 64 + j] * vdec + vmsg[((long)r * 3 + l) * 64 + j];
  }
  vembT[idx] = f2bf(v);
}

// ------------- hh(2,N,64) -> transpose(1,0,2) -> stA(N,128) bf16 -----------
__global__ void k_remap(const ushort_t* __restrict__ hh, ushort_t* stA, int N)
{
  long idx = (long)blockIdx.x * blockDim.x + threadIdx.x;
  if (idx >= (long)N * 128) return;
  int n = (int)(idx >> 7), c = (int)(idx & 127);
  stA[idx] = hh[(long)((c >> 6) ? (N + n) : n) * 64 + (c & 63)];
}

// ------------- final blend -------------------------------------------------
__global__ void k_final(const float* __restrict__ spat,
                        const float* __restrict__ stat,
                        const float* lamb, float* out, long total)
{
  long idx = (long)blockIdx.x * blockDim.x + threadIdx.x;
  if (idx >= total) return;
  float lm = lamb[0];
  out[idx] = lm * stat[idx] + (1.f - lm) * spat[idx];
}

// ------------- weight f32 (K x Nw) -> transposed bf16 (Nw x K) -------------
__global__ void k_wt(const float* __restrict__ in, ushort_t* out, int K, int Nw)
{
  int idx = blockIdx.x * blockDim.x + threadIdx.x;
  if (idx >= K * Nw) return;
  int k = idx / Nw, j = idx % Nw;
  out[(long)j * K + k] = f2bf(in[idx]);
}

extern "C" void kernel_launch(void* const* d_in, const int* in_sizes, int n_in,
                              void* d_out, int out_size, void* d_ws, size_t ws_size,
                              hipStream_t stream)
{
  const int*   src    = (const int*)d_in[0];
  const int*   tgt    = (const int*)d_in[1];
  const float* ts     = (const float*)d_in[2];
  const float* now    = (const float*)d_in[3];
  const float* nfeat  = (const float*)d_in[5];
  const float* memory = (const float*)d_in[6];
  const float* lastup = (const float*)d_in[7];
  const float* vmem   = (const float*)d_in[8];
  const float* vlu    = (const float*)d_in[9];
  const float* A_r    = (const float*)d_in[10];
  const float* stat   = (const float*)d_in[11];
  const float* lamb   = (const float*)d_in[12];
  const float* lambs  = (const float*)d_in[13];
  const float* mf_W1  = (const float*)d_in[14];
  const float* mf_b1  = (const float*)d_in[15];
  const float* mf_W2  = (const float*)d_in[16];
  const float* mf_b2  = (const float*)d_in[17];
  const float* ffr_W1 = (const float*)d_in[18];
  const float* ffr_b1 = (const float*)d_in[19];
  const float* ffr_W2 = (const float*)d_in[20];
  const float* ffr_b2 = (const float*)d_in[21];
  const float* et_W   = (const float*)d_in[22];
  const float* et_b   = (const float*)d_in[23];
  const float* st_W   = (const float*)d_in[24];
  const float* st_b   = (const float*)d_in[25];

  const int E = in_sizes[0];
  const int N = in_sizes[7];            // last_update is (N,)
  const int R = in_sizes[9];            // virtual_last_update is (R,)
  const int Npad = (N + 31) & ~31;
  const int Rpad = (R + 31) & ~31;
  const int KCH3 = 8;                   // split-K factor for the big GEMM

  char* wsp = (char*)d_ws;
  size_t off = 0;
  auto alloc = [&](size_t bytes) -> void* {
    void* p = wsp + off;
    off += (bytes + 255) & ~(size_t)255;
    return p;
  };

  unsigned int* tmax_bits = (unsigned int*)alloc((size_t)N * 4);
  int*      cnt      = (int*)     alloc((size_t)N * 4);
  float*    agg      = (float*)   alloc((size_t)N * 3 * 257 * 4);
  float*    vmid_f32 = (float*)   alloc((size_t)R * 768 * 4);
  size_t zero_bytes = off;                          // zero tmax+cnt+agg+vmid
  ushort_t* h_bf     = (ushort_t*)alloc((size_t)3 * N * 128 * 2);
  float*    msg_feat = (float*)   alloc((size_t)3 * N * 64 * 4);
  ushort_t* Bt_big   = (ushort_t*)alloc((size_t)768 * Npad * 2);
  ushort_t* At       = (ushort_t*)alloc((size_t)R * Npad * 2);
  ushort_t* r2n      = (ushort_t*)alloc((size_t)N * Rpad * 2);
  ushort_t* vmid_bf  = (ushort_t*)alloc((size_t)R * 768 * 2);
  ushort_t* vh_bf    = (ushort_t*)alloc((size_t)R * 3 * 256 * 2);
  float*    vmsg     = (float*)   alloc((size_t)R * 3 * 64 * 4);
  ushort_t* vembT    = (ushort_t*)alloc((size_t)192 * Rpad * 2);
  ushort_t* dyn_bf   = (ushort_t*)alloc((size_t)2 * N * 192 * 2);
  ushort_t* hh_bf    = (ushort_t*)alloc((size_t)2 * N * 64 * 2);
  ushort_t* stA      = (ushort_t*)alloc((size_t)N * 128 * 2);
  float*    spat     = (float*)   alloc((size_t)N * 128 * 4);
  ushort_t* w1t = (ushort_t*)alloc(128 * 256 * 2);
  ushort_t* w2t = (ushort_t*)alloc(64 * 128 * 2);
  ushort_t* f1t = (ushort_t*)alloc(256 * 256 * 2);
  ushort_t* f2t = (ushort_t*)alloc(64 * 256 * 2);
  ushort_t* ett = (ushort_t*)alloc(64 * 192 * 2);
  ushort_t* stt = (ushort_t*)alloc(128 * 128 * 2);
  (void)ws_size; (void)n_in; (void)out_size;

  auto cdiv = [](long a, long b) { return (int)((a + b - 1) / b); };
  const int TB = 256;

  (void)hipMemsetAsync(d_ws, 0, zero_bytes, stream);

  k_seg<<<cdiv(E, TB), TB, 0, stream>>>(src, ts, tmax_bits, cnt, E);

  k_wt<<<cdiv(256 * 128, TB), TB, 0, stream>>>(mf_W1, w1t, 256, 128);
  k_wt<<<cdiv(128 * 64,  TB), TB, 0, stream>>>(mf_W2, w2t, 128, 64);
  k_wt<<<cdiv(256 * 256, TB), TB, 0, stream>>>(ffr_W1, f1t, 256, 256);
  k_wt<<<cdiv(256 * 64,  TB), TB, 0, stream>>>(ffr_W2, f2t, 256, 64);
  k_wt<<<cdiv(192 * 64,  TB), TB, 0, stream>>>(et_W, ett, 192, 64);
  k_wt<<<cdiv(128 * 128, TB), TB, 0, stream>>>(st_W, stt, 128, 128);

  k_edge_scatter<<<cdiv((long)E * 32, TB), TB, 0, stream>>>(
      src, tgt, ts, nfeat, memory, tmax_bits, lambs, agg, E);

  k_col_softmax<<<R, 256, 0, stream>>>(A_r, cnt, At, N, R, Npad);
  k_row_softmax<<<cdiv((long)N * 32, TB), TB, 0, stream>>>(A_r, r2n, N, R, Rpad);

  // G1: h = relu(feat @ mf_W1 + b1)   M=3N, K=256 (f32 A from agg, lda=257)
  { int Mt = (3 * N) / 16, Nt = 128 / 64;
    gemm_bf16_wmma<1,1,1><<<cdiv((long)Mt * Nt * 32, TB), TB, 0, stream>>>(
        nullptr, agg, 257, w1t, 256, mf_b1, nullptr, h_bf, Mt, Nt, 256, 128, 1); }
  // G2: msg_feat = relu(h @ mf_W2 + b2)   M=3N, K=128, N=64
  { int Mt = (3 * N) / 16, Nt = 1;
    gemm_bf16_wmma<0,1,0><<<cdiv((long)Mt * Nt * 32, TB), TB, 0, stream>>>(
        h_bf, nullptr, 128, w2t, 128, mf_b2, msg_feat, nullptr, Mt, Nt, 128, 64, 1); }

  k_mem_update<<<cdiv((long)3 * N, TB), TB, 0, stream>>>(
      memory, agg, msg_feat, tmax_bits, cnt, lastup, lambs, dyn_bf, N);
  k_postB<<<cdiv((long)768 * Npad, TB), TB, 0, stream>>>(agg, cnt, Bt_big, N, Npad);

  // G3: vmid = A^T @ B   M=R, K=Npad, N=768, split-K with f32 atomic epilogue
  { int Mt = R / 16, Nt = 768 / 64;
    dim3 grid(cdiv((long)Mt * Nt * 32, TB), KCH3);
    gemm_bf16_wmma<0,0,2><<<grid, TB, 0, stream>>>(
        At, nullptr, Npad, Bt_big, Npad, nullptr, vmid_f32, nullptr,
        Mt, Nt, Npad, 768, KCH3); }
  k_cvt<<<cdiv((long)R * 768, TB), TB, 0, stream>>>(vmid_f32, vmid_bf, (long)R * 768);

  // G4: vh = leaky(vmid @ ffr_W1 + b1)   M=3R, K=256, N=256
  { int Mt = (3 * R) / 16, Nt = 256 / 64;
    gemm_bf16_wmma<0,2,1><<<cdiv((long)Mt * Nt * 32, TB), TB, 0, stream>>>(
        vmid_bf, nullptr, 256, f1t, 256, ffr_b1, nullptr, vh_bf, Mt, Nt, 256, 256, 1); }
  // G5: vmsg = leaky(vh @ ffr_W2 + b2)   M=3R, K=256, N=64
  { int Mt = (3 * R) / 16, Nt = 1;
    gemm_bf16_wmma<0,2,0><<<cdiv((long)Mt * Nt * 32, TB), TB, 0, stream>>>(
        vh_bf, nullptr, 256, f2t, 256, ffr_b2, vmsg, nullptr, Mt, Nt, 256, 64, 1); }

  k_vupdate<<<cdiv((long)192 * Rpad, TB), TB, 0, stream>>>(
      vmem, vlu, now, lambs, vmsg, vembT, R, Rpad);

  // G6: vnode_emb = r2n @ v_emb -> dyn rows [N, 2N)   M=N, K=Rpad, N=192
  { int Mt = N / 16, Nt = 192 / 64;
    gemm_bf16_wmma<0,0,1><<<cdiv((long)Mt * Nt * 32, TB), TB, 0, stream>>>(
        r2n, nullptr, Rpad, vembT, Rpad, nullptr, nullptr,
        dyn_bf + (size_t)N * 192, Mt, Nt, Rpad, 192, 1); }
  // G7: hh = leaky(dyn @ et_W + b)   M=2N, K=192, N=64
  { int Mt = (2 * N) / 16, Nt = 1;
    gemm_bf16_wmma<0,2,1><<<cdiv((long)Mt * Nt * 32, TB), TB, 0, stream>>>(
        dyn_bf, nullptr, 192, ett, 192, et_b, nullptr, hh_bf, Mt, Nt, 192, 64, 1); }

  k_remap<<<cdiv((long)N * 128, TB), TB, 0, stream>>>(hh_bf, stA, N);

  // G8: spat = leaky(stA @ st_W + b)   M=N, K=128, N=128
  { int Mt = N / 16, Nt = 128 / 64;
    gemm_bf16_wmma<0,2,0><<<cdiv((long)Mt * Nt * 32, TB), TB, 0, stream>>>(
        stA, nullptr, 128, stt, 128, st_b, spat, nullptr, Mt, Nt, 128, 128, 1); }

  k_final<<<cdiv((long)N * 128, TB), TB, 0, stream>>>(
      spat, stat, lamb, (float*)d_out, (long)N * 128);
}
